// TransformerSenderReinforce_47545287967445
// MI455X (gfx1250) — matrix operand order repaired and checked by reading, not compiled, not run on hardware
//
#include <hip/hip_runtime.h>
#include <hip/hip_bf16.h>
#include <math.h>

typedef _Float16 h16;
typedef __attribute__((ext_vector_type(16))) h16   v16h;
typedef __attribute__((ext_vector_type(8)))  h16   v8h;
typedef __attribute__((ext_vector_type(8)))  float v8f;

#define B_  32
#define V_  32000
#define D_  512
#define H_  8
#define F_  2048
#define NL_ 2
#define L_  12
#define DH_ 64
#define ML_ (B_ * L_)   // 384 rows processed per decoder pass

// ---------------------------------------------------------------------------
// f32 -> f16 plain conversion (A-side activations)
// ---------------------------------------------------------------------------
__global__ void cvt_kernel(const float* __restrict__ in, h16* __restrict__ out, int n) {
  for (int i = blockIdx.x * blockDim.x + threadIdx.x; i < n; i += gridDim.x * blockDim.x)
    out[i] = (h16)in[i];
}

// ---------------------------------------------------------------------------
// Fused f32 -> f16 convert + transpose: in (K x N) -> out (N x K).
// Transposed weights make the WMMA B-fragment a contiguous 32B load per lane
// (two global_load_b128) instead of 16 strided global_load_u16.
// One pass per weight per launch; weights are L2-resident afterward.
// ---------------------------------------------------------------------------
__global__ __launch_bounds__(256) void cvtT_kernel(const float* __restrict__ in,
    h16* __restrict__ out, int K, int N) {
  __shared__ h16 tile[32][33];                       // +1 pad kills bank conflicts
  int kb = blockIdx.x * 32, nb = blockIdx.y * 32;
  int tx = threadIdx.x & 31, ty = threadIdx.x >> 5;  // 32 x 8
#pragma unroll
  for (int r = ty; r < 32; r += 8)                   // coalesced read of rows k
    tile[r][tx] = (h16)in[(size_t)(kb + r) * N + (nb + tx)];
  __syncthreads();
#pragma unroll
  for (int r = ty; r < 32; r += 8)                   // coalesced write of rows n
    out[(size_t)(nb + r) * K + (kb + tx)] = tile[tx][r];
}

// ---------------------------------------------------------------------------
// Sinusoidal positional encoding, computed on device (not a harness input)
// ---------------------------------------------------------------------------
__global__ void posinit_kernel(float* __restrict__ pos) {
  int i = blockIdx.x * 256 + threadIdx.x;            // L_*D_ = 6144 threads
  int p = i / D_, d = i % D_;
  int ii = d >> 1;
  float ang = (float)p * __expf(-(2.0f * (float)ii / (float)D_) * 9.210340371976184f); // ln 1e4
  pos[i] = (d & 1) ? __cosf(ang) : __sinf(ang);
}

__global__ void bufinit_kernel(float* __restrict__ buf, const float* __restrict__ sym) {
  int i = blockIdx.x * 256 + threadIdx.x;            // B_*L_*D_
  int p = (i / D_) % L_;
  buf[i] = (p == 0) ? sym[i % D_] : 0.0f;
}

__global__ void addpos_kernel(const float* __restrict__ buf, const float* __restrict__ pos,
                              float* __restrict__ h) {
  int i = blockIdx.x * 256 + threadIdx.x;
  h[i] = buf[i] + pos[i % (L_ * D_)];
}

// h[b,p,:] += crossAdd[b,:]  (cross-attention block collapses to a constant)
__global__ void addcross_kernel(float* __restrict__ h, const float* __restrict__ cr) {
  int i = blockIdx.x * 256 + threadIdx.x;
  int b = i / (L_ * D_);
  int d = i % D_;
  h[i] += cr[b * D_ + d];
}

// ---------------------------------------------------------------------------
// LayerNorm: one 256-thread block per row (D=512, 2 elems/thread), f16 output
// ---------------------------------------------------------------------------
__global__ __launch_bounds__(256) void ln_kernel(const float* __restrict__ x, int xstride,
    const float* __restrict__ g, const float* __restrict__ b, h16* __restrict__ out) {
  int row = blockIdx.x, tid = threadIdx.x;
  const float* xr = x + (size_t)row * (size_t)xstride;
  float v0 = xr[tid], v1 = xr[tid + 256];
  __shared__ float sh[10];
  float s = v0 + v1;
  for (int o = 16; o; o >>= 1) s += __shfl_xor(s, o, 32);
  int w = tid >> 5;
  if ((tid & 31) == 0) sh[w] = s;
  __syncthreads();
  if (tid == 0) { float t = 0.f; for (int i = 0; i < 8; ++i) t += sh[i]; sh[8] = t * (1.0f / D_); }
  __syncthreads();
  float mean = sh[8];
  float d0 = v0 - mean, d1 = v1 - mean;
  float q = d0 * d0 + d1 * d1;
  for (int o = 16; o; o >>= 1) q += __shfl_xor(q, o, 32);
  __syncthreads();
  if ((tid & 31) == 0) sh[w] = q;
  __syncthreads();
  if (tid == 0) { float t = 0.f; for (int i = 0; i < 8; ++i) t += sh[i]; sh[9] = rsqrtf(t * (1.0f / D_) + 1e-5f); }
  __syncthreads();
  float rs = sh[9];
  size_t ob = (size_t)row * D_;
  out[ob + tid]       = (h16)(d0 * rs * g[tid] + b[tid]);
  out[ob + tid + 256] = (h16)(d1 * rs * g[tid + 256] + b[tid + 256]);
}

// ---------------------------------------------------------------------------
// WMMA GEMM, B pre-transposed (N x K). One wave -> 16x32 output (two 16x16
// tiles sharing one A fragment -> two independent v_wmma per K step, ILP
// across the accumulate chain). Fragments per CDNA5 ISA 7.12.2.
// MODE 0: C_f32 = acc+bias   MODE 1: C_f32 += acc+bias (residual)
// MODE 2: C_f16 = relu(acc+bias)   MODE 3: C_f16 = acc+bias
// ---------------------------------------------------------------------------
template <int MODE>
__global__ __launch_bounds__(256) void gemm_wmma(const h16* __restrict__ A,
    const h16* __restrict__ Bt, const float* __restrict__ bias,
    float* __restrict__ Cf, h16* __restrict__ Ch, int M, int N, int K) {
  const int tilesN2 = N >> 5;                         // pairs of 16-wide N tiles
  const int totalTiles = (M >> 4) * tilesN2;
  const int wave = (int)((blockIdx.x * blockDim.x + threadIdx.x) >> 5);
  if (wave >= totalTiles) return;                     // whole-wave uniform exit
  const int tM = wave / tilesN2, tN2 = wave % tilesN2;
  const int lane = threadIdx.x & 31;
  const int hi = lane >> 4;                           // lane half selects K sub-block
  const int lo = lane & 15;

  const h16* Arow  = A  + (size_t)(tM * 16 + lo) * (size_t)K;          // A: m = lane%16
  const h16* Brow0 = Bt + (size_t)(tN2 * 32 + lo) * (size_t)K;         // n0 = base+lo
  const h16* Brow1 = Bt + (size_t)(tN2 * 32 + 16 + lo) * (size_t)K;    // n1 = n0+16
  v8f acc0 = {}, acc1 = {};
  for (int k0 = 0; k0 < K; k0 += 32) {
    // A fragment: 16 consecutive K-values K[hi*16 .. +15] (matches B)
    v16h a = *(const v16h*)(Arow + k0 + hi * 16);
    // B fragment: 16 consecutive K-values of column n -> contiguous 32B load
    v16h b0 = *(const v16h*)(Brow0 + k0 + hi * 16);
    v16h b1 = *(const v16h*)(Brow1 + k0 + hi * 16);
    if (k0 + 32 < K) {
      __builtin_prefetch(Brow0 + k0 + 32, 0, 1);
      __builtin_prefetch(Brow1 + k0 + 32, 0, 1);
    }
    acc0 = __builtin_amdgcn_wmma_f32_16x16x32_f16(false, a, false, b0, (short)0, acc0,
                                                  false, false);
    acc1 = __builtin_amdgcn_wmma_f32_16x16x32_f16(false, a, false, b1, (short)0, acc1,
                                                  false, false);
  }
  const int n0 = tN2 * 32 + lo;
  const float bv0 = bias ? bias[n0] : 0.0f;
  const float bv1 = bias ? bias[n0 + 16] : 0.0f;
#pragma unroll
  for (int r = 0; r < 8; ++r) {                       // C: m = 8*hi + r, n = lane%16
    const int m = tM * 16 + hi * 8 + r;
    float val0 = acc0[r] + bv0;
    float val1 = acc1[r] + bv1;
    if (MODE == 0)      { Cf[(size_t)m * N + n0]  = val0; Cf[(size_t)m * N + n0 + 16]  = val1; }
    else if (MODE == 1) { Cf[(size_t)m * N + n0] += val0; Cf[(size_t)m * N + n0 + 16] += val1; }
    else if (MODE == 2) { Ch[(size_t)m * N + n0]  = (h16)fmaxf(val0, 0.0f);
                          Ch[(size_t)m * N + n0 + 16] = (h16)fmaxf(val1, 0.0f); }
    else                { Ch[(size_t)m * N + n0]  = (h16)val0;
                          Ch[(size_t)m * N + n0 + 16] = (h16)val1; }
  }
}

// ---------------------------------------------------------------------------
// Self-attention: one wave per (b, head, query-pos). Keys 0..t (mask is
// per-step, not per-query). Shuffle softmax; f16 output feeds the Wo GEMM.
// ---------------------------------------------------------------------------
__global__ __launch_bounds__(256) void attn_kernel(const float* __restrict__ q,
    const float* __restrict__ k, const float* __restrict__ v,
    h16* __restrict__ oh, int t) {
  int wid  = (int)((blockIdx.x * blockDim.x + threadIdx.x) >> 5);
  int lane = threadIdx.x & 31;
  if (wid >= B_ * H_ * L_) return;
  int p  = wid % L_;
  int hh = (wid / L_) % H_;
  int b  = wid / (L_ * H_);
  const float* qp = q + ((size_t)(b * L_ + p) * D_) + hh * DH_;
  float s = -1.0e9f;
  if (lane <= t) {
    const float* kp = k + ((size_t)(b * L_ + lane) * D_) + hh * DH_;
    float acc = 0.f;
#pragma unroll 16
    for (int d = 0; d < DH_; ++d) acc += qp[d] * kp[d];
    s = acc * 0.125f;                                // 1/sqrt(64)
  }
  float m = s;
  for (int o = 16; o; o >>= 1) m = fmaxf(m, __shfl_xor(m, o, 32));
  float e = (lane <= t) ? __expf(s - m) : 0.0f;
  float sum = e;
  for (int o = 16; o; o >>= 1) sum += __shfl_xor(sum, o, 32);
  float a = e / sum;
  float o0 = 0.f, o1 = 0.f;
  for (int kk = 0; kk <= t; ++kk) {
    float ak = __shfl(a, kk, 32);
    const float* vp = v + ((size_t)(b * L_ + kk) * D_) + hh * DH_;
    o0 += ak * vp[lane];
    o1 += ak * vp[lane + 32];
  }
  h16* op = oh + ((size_t)(b * L_ + p) * D_) + hh * DH_;
  op[lane]      = (h16)o0;
  op[lane + 32] = (h16)o1;
}

// ---------------------------------------------------------------------------
// Per-batch argmax + log-softmax + entropy over V=32000, then write the next
// token embedding into buf (fused to stay off the critical launch path).
// Outputs: seq(B,L) | logps(B,L) | ents(B,L) flattened into d_out as float.
// ---------------------------------------------------------------------------
__global__ __launch_bounds__(256) void sample_kernel(const float* __restrict__ logits,
    const float* __restrict__ emb, float* __restrict__ buf,
    float* __restrict__ out, int t) {
  __shared__ float sv[256];
  __shared__ int   si[256];
  __shared__ float s2[256];
  int b = blockIdx.x, tid = threadIdx.x;
  const float* lg = logits + (size_t)b * V_;
  float mx = -3.4e38f; int ai = 0;
  for (int i = tid; i < V_; i += 256) {
    float x = lg[i];
    if (x > mx || (x == mx && i < ai)) { mx = x; ai = i; }
  }
  sv[tid] = mx; si[tid] = ai;
  __syncthreads();
  for (int st = 128; st > 0; st >>= 1) {
    if (tid < st) {
      float ov = sv[tid + st]; int oi = si[tid + st];
      if (ov > sv[tid] || (ov == sv[tid] && oi < si[tid])) { sv[tid] = ov; si[tid] = oi; }
    }
    __syncthreads();
  }
  mx = sv[0]; ai = si[0];
  __syncthreads();
  float se = 0.f, swe = 0.f;
  for (int i = tid; i < V_; i += 256) {
    float d = lg[i] - mx;
    float e = __expf(d);
    se += e; swe += e * d;
  }
  sv[tid] = se; s2[tid] = swe;
  __syncthreads();
  for (int st = 128; st > 0; st >>= 1) {
    if (tid < st) { sv[tid] += sv[tid + st]; s2[tid] += s2[tid + st]; }
    __syncthreads();
  }
  if (tid == 0) {
    float Z = sv[0], logZ = __logf(Z);
    out[b * L_ + t]               = (float)ai;        // seq
    out[B_ * L_ + b * L_ + t]     = -logZ;            // logp of argmax (logit==max)
    out[2 * B_ * L_ + b * L_ + t] = logZ - s2[0] / Z; // entropy
  }
  if (t + 1 < L_) {                                   // buf.at[:, t+1].set (mode='drop')
    const float* er = emb + (size_t)ai * D_;
    float* br = buf + ((size_t)b * L_ + (size_t)(t + 1)) * D_;
    for (int d = tid; d < D_; d += 256) br[d] = er[d] * 22.627416997969522f; // sqrt(512)
  }
}

// ---------------------------------------------------------------------------
extern "C" void kernel_launch(void* const* d_in, const int* in_sizes, int n_in,
                              void* d_out, int out_size, void* d_ws, size_t ws_size,
                              hipStream_t stream) {
  (void)in_sizes; (void)n_in; (void)out_size; (void)ws_size;
  const float* x    = (const float*)d_in[0];
  const float* Wq   = (const float*)d_in[1];
  const float* bq   = (const float*)d_in[2];
  const float* Wk   = (const float*)d_in[3];
  const float* bk   = (const float*)d_in[4];
  const float* Wv   = (const float*)d_in[5];
  const float* bv   = (const float*)d_in[6];
  const float* Wo   = (const float*)d_in[7];
  const float* bo   = (const float*)d_in[8];
  /* Cq(9)/cbq(10)/Ck(11)/cbk(12) unused: cross-softmax over 1 key == 1 */
  const float* Cv   = (const float*)d_in[13];
  const float* cbv  = (const float*)d_in[14];
  const float* Co   = (const float*)d_in[15];
  const float* cbo  = (const float*)d_in[16];
  const float* W1   = (const float*)d_in[17];
  const float* b1   = (const float*)d_in[18];
  const float* W2   = (const float*)d_in[19];
  const float* b2   = (const float*)d_in[20];
  const float* ln1g = (const float*)d_in[21];
  const float* ln1b = (const float*)d_in[22];
  /* ln2(23,24) unused for the same reason */
  const float* ln3g = (const float*)d_in[25];
  const float* ln3b = (const float*)d_in[26];
  const float* emb  = (const float*)d_in[27];
  const float* ssym = (const float*)d_in[28];
  const float* Wout = (const float*)d_in[29];
  const float* bout = (const float*)d_in[30];
  const float* lnfg = (const float*)d_in[31];
  const float* lnfb = (const float*)d_in[32];

  // ---- workspace carve (bump allocator, 256B aligned) ----
  char* p = (char*)d_ws;
  auto takeH = [&](size_t n) { h16* r = (h16*)p;  p += ((n * sizeof(h16)  + 255) & ~(size_t)255); return r; };
  auto takeF = [&](size_t n) { float* r = (float*)p; p += ((n * sizeof(float) + 255) & ~(size_t)255); return r; };

  h16* WqT   = takeH((size_t)NL_ * D_ * D_);   // all weight copies are transposed (N x K), f16
  h16* WkT   = takeH((size_t)NL_ * D_ * D_);
  h16* WvT   = takeH((size_t)NL_ * D_ * D_);
  h16* WoT   = takeH((size_t)NL_ * D_ * D_);
  h16* CvT   = takeH((size_t)NL_ * D_ * D_);
  h16* CoT   = takeH((size_t)NL_ * D_ * D_);
  h16* W1T   = takeH((size_t)NL_ * D_ * F_);
  h16* W2T   = takeH((size_t)NL_ * F_ * D_);
  h16* WoutT = takeH((size_t)D_ * V_);
  h16* xh    = takeH((size_t)B_ * D_);
  h16* t1h   = takeH((size_t)B_ * D_);
  h16* zhh   = takeH((size_t)ML_ * D_);
  h16* ohh   = takeH((size_t)ML_ * D_);
  h16* ff1h  = takeH((size_t)ML_ * F_);
  h16* zfh   = takeH((size_t)B_ * D_);
  float* POSd   = takeF((size_t)L_ * D_);
  float* buf    = takeF((size_t)ML_ * D_);
  float* hb     = takeF((size_t)ML_ * D_);
  float* qb     = takeF((size_t)ML_ * D_);
  float* kb     = takeF((size_t)ML_ * D_);
  float* vb     = takeF((size_t)ML_ * D_);
  float* crossA = takeF((size_t)NL_ * B_ * D_);
  float* logits = takeF((size_t)B_ * V_);

  auto cvt = [&](const float* in, h16* o, int n) {
    int blocks = (n + 255) / 256; if (blocks > 8192) blocks = 8192;
    cvt_kernel<<<blocks, 256, 0, stream>>>(in, o, n);
  };
  auto cvtT = [&](const float* in, h16* o, int K, int N) {   // in (KxN) -> out (NxK)
    dim3 grid(K / 32, N / 32);
    cvtT_kernel<<<grid, 256, 0, stream>>>(in, o, K, N);
  };
  auto gemm = [&](int mode, const h16* A, const h16* Bt, const float* bias,
                  float* Cf, h16* Ch, int M, int N, int K) {
    int tiles = (M / 16) * (N / 32);                 // one wave per 16x32 output
    int blocks = (tiles + 7) / 8;                    // 8 waves / 256-thread block
    switch (mode) {
      case 0: gemm_wmma<0><<<blocks, 256, 0, stream>>>(A, Bt, bias, Cf, Ch, M, N, K); break;
      case 1: gemm_wmma<1><<<blocks, 256, 0, stream>>>(A, Bt, bias, Cf, Ch, M, N, K); break;
      case 2: gemm_wmma<2><<<blocks, 256, 0, stream>>>(A, Bt, bias, Cf, Ch, M, N, K); break;
      default: gemm_wmma<3><<<blocks, 256, 0, stream>>>(A, Bt, bias, Cf, Ch, M, N, K); break;
    }
  };

  // ---- once per launch: weights -> f16 transposed, POS, buf init, cross const ----
  for (int l = 0; l < NL_; ++l) {
    const int dd = l * D_ * D_, df = l * D_ * F_;
    cvtT(Wq + dd, WqT + dd, D_, D_);
    cvtT(Wk + dd, WkT + dd, D_, D_);
    cvtT(Wv + dd, WvT + dd, D_, D_);
    cvtT(Wo + dd, WoT + dd, D_, D_);
    cvtT(Cv + dd, CvT + dd, D_, D_);
    cvtT(Co + dd, CoT + dd, D_, D_);
    cvtT(W1 + df, W1T + df, D_, F_);
    cvtT(W2 + df, W2T + df, F_, D_);
  }
  cvtT(Wout, WoutT, D_, V_);
  cvt(x, xh, B_ * D_);
  posinit_kernel<<<(L_ * D_) / 256, 256, 0, stream>>>(POSd);
  bufinit_kernel<<<(ML_ * D_) / 256, 256, 0, stream>>>(buf, ssym);
  for (int l = 0; l < NL_; ++l) {
    gemm(3, xh,  CvT + l * D_ * D_, cbv + l * D_, nullptr, t1h, B_, D_, D_);
    gemm(0, t1h, CoT + l * D_ * D_, cbo + l * D_, crossA + l * B_ * D_, nullptr, B_, D_, D_);
  }

  // ---- 12 dependent decode steps ----
  for (int t = 0; t < L_; ++t) {
    addpos_kernel<<<(ML_ * D_) / 256, 256, 0, stream>>>(buf, POSd, hb);
    for (int l = 0; l < NL_; ++l) {
      const int wDD = l * D_ * D_, wD = l * D_;
      ln_kernel<<<ML_, 256, 0, stream>>>(hb, D_, ln1g + wD, ln1b + wD, zhh);
      gemm(0, zhh, WqT + wDD, bq + wD, qb, nullptr, ML_, D_, D_);
      gemm(0, zhh, WkT + wDD, bk + wD, kb, nullptr, ML_, D_, D_);
      gemm(0, zhh, WvT + wDD, bv + wD, vb, nullptr, ML_, D_, D_);
      attn_kernel<<<(B_ * H_ * L_ * 32) / 256, 256, 0, stream>>>(qb, kb, vb, ohh, t);
      gemm(1, ohh, WoT + wDD, bo + wD, hb, nullptr, ML_, D_, D_);     // h += attn@Wo+bo
      addcross_kernel<<<(ML_ * D_) / 256, 256, 0, stream>>>(hb, crossA + l * B_ * D_);
      ln_kernel<<<ML_, 256, 0, stream>>>(hb, D_, ln3g + wD, ln3b + wD, zhh);
      gemm(2, zhh, W1T + l * D_ * F_, b1 + l * F_, nullptr, ff1h, ML_, F_, D_); // relu->f16
      gemm(1, ff1h, W2T + l * F_ * D_, b2 + wD, hb, nullptr, ML_, D_, F_);      // h += ffn
    }
    ln_kernel<<<B_, 256, 0, stream>>>(hb + t * D_, L_ * D_, lnfg, lnfb, zfh);   // row t only
    gemm(0, zfh, WoutT, bout, logits, nullptr, B_, V_, D_);
    sample_kernel<<<B_, 256, 0, stream>>>(logits, emb, buf, (float*)d_out, t);
  }
}
